// Net_Fuse_83657372991510
// MI455X (gfx1250) — matrix operand order repaired and checked by reading, not compiled, not run on hardware
//
#include <hip/hip_runtime.h>
#include <hip/hip_bf16.h>

// ===========================================================================
// MI455X (gfx1250) implementation of the MEE-style multimodal fusion forward.
//  * Collapse each GU's fc1+fc2 (no activation between) into W12 = w2@w1 once
//    per call (small fp32 GEMM), cutting the dominant GEMM FLOPs ~4x.
//  * All large GEMMs (per-modality x@W12^T, context-gating, and the final
//    4096x4096 distance matrix) run through a single wave32 WMMA bf16 kernel
//    (v_wmma_f32_16x16x32_bf16, fp32 accumulate), 128x64 block tiles,
//    8 waves x 32x32, double-buffered LDS, async global->LDS staging when the
//    toolchain exposes the gfx1250 async-to-LDS builtins.
//  * NetVLAD / BN / softmax / gating / L2 norms stay fp32 on VALU (tiny cost).
// ===========================================================================

typedef __bf16 bf16_t;
typedef __attribute__((ext_vector_type(16))) __bf16 v16bf;
typedef __attribute__((ext_vector_type(8)))  __bf16 v8bf;
typedef __attribute__((ext_vector_type(8)))  float  v8f;
typedef int v4i __attribute__((vector_size(16)));  // 128-bit async payload

#define NB 4096
#define EPSN 1e-12f

#if defined(__has_builtin)
#if __has_builtin(__builtin_amdgcn_global_load_async_to_lds_b128) && \
    __has_builtin(__builtin_amdgcn_s_wait_asynccnt)
#define USE_ASYNC_LDS 1
#endif
#endif
#ifndef USE_ASYNC_LDS
#define USE_ASYNC_LDS 0
#endif

#if USE_ASYNC_LDS
#define AS1 __attribute__((address_space(1)))
#define AS3 __attribute__((address_space(3)))
#endif

// --------------------------------------------------------------------------
// f32 -> bf16 elementwise convert
// --------------------------------------------------------------------------
__global__ __launch_bounds__(256) void k_cvt_bf16(const float* __restrict__ in,
                                                  bf16_t* __restrict__ out,
                                                  long n) {
  long i = (long)blockIdx.x * blockDim.x + threadIdx.x;
  long stride = (long)gridDim.x * blockDim.x;
  for (; i < n; i += stride) out[i] = (bf16_t)in[i];
}

// --------------------------------------------------------------------------
// Fused NetVLAD: one workgroup per batch element, fp32 in LDS, bf16 out.
// x:(B,T,D) clusters:(D,K) clusters2:(D,K) bn over K. out:(B,D*K) bf16.
// Max sizes: T=30, D=300, K=32  ->  ~80KB LDS (well under 320KB/WGP).
// --------------------------------------------------------------------------
__global__ __launch_bounds__(256) void k_netvlad(
    const float* __restrict__ x, const float* __restrict__ clusters,
    const float* __restrict__ clusters2,
    const float* __restrict__ bnb, const float* __restrict__ bnm,
    const float* __restrict__ bns, const float* __restrict__ bnv,
    bf16_t* __restrict__ out, int T, int D, int K) {
  __shared__ float xs[9000];    // T*D
  __shared__ float as[960];     // T*K  (assignments)
  __shared__ float csum[32];    // per-k a_sum, later per-k inv-norm
  __shared__ float vlad[9600];  // D*K
  __shared__ float red[256];
  const int b = blockIdx.x, tid = threadIdx.x;
  const float* xb = x + (size_t)b * T * D;

  for (int i = tid; i < T * D; i += 256) xs[i] = xb[i];
  __syncthreads();

  // assignments: bn(x @ clusters)
  for (int i = tid; i < T * K; i += 256) {
    int t = i / K, k = i % K;
    float s = 0.f;
    for (int d = 0; d < D; ++d) s += xs[t * D + d] * clusters[d * K + k];
    as[i] = (s - bnm[k]) * rsqrtf(bnv[k] + 1e-5f) * bns[k] + bnb[k];
  }
  __syncthreads();

  // softmax over K per time-step
  if (tid < T) {
    float mx = -3.0e38f;
    for (int k = 0; k < K; ++k) mx = fmaxf(mx, as[tid * K + k]);
    float sm = 0.f;
    for (int k = 0; k < K; ++k) {
      float e = __expf(as[tid * K + k] - mx);
      as[tid * K + k] = e;
      sm += e;
    }
    float inv = 1.f / sm;
    for (int k = 0; k < K; ++k) as[tid * K + k] *= inv;
  }
  __syncthreads();

  if (tid < K) {
    float s = 0.f;
    for (int t = 0; t < T; ++t) s += as[t * K + tid];
    csum[tid] = s;
  }
  __syncthreads();

  // vlad[d,k] = sum_t a[t,k]*x[t,d] - a_sum[k]*clusters2[d,k]
  for (int i = tid; i < D * K; i += 256) {
    int d = i / K, k = i % K;
    float s = 0.f;
    for (int t = 0; t < T; ++t) s += as[t * K + k] * xs[t * D + d];
    vlad[i] = s - csum[k] * clusters2[i];
  }
  __syncthreads();

  // intra-norm over D per cluster column
  if (tid < K) {
    float ss = 0.f;
    for (int d = 0; d < D; ++d) {
      float v = vlad[d * K + tid];
      ss += v * v;
    }
    csum[tid] = 1.f / fmaxf(sqrtf(ss), EPSN);
  }
  __syncthreads();

  float part = 0.f;
  for (int i = tid; i < D * K; i += 256) {
    float v = vlad[i] * csum[i % K];
    vlad[i] = v;
    part += v * v;
  }
  red[tid] = part;
  __syncthreads();
  for (int s = 128; s > 0; s >>= 1) {
    if (tid < s) red[tid] += red[tid + s];
    __syncthreads();
  }
  float rn = 1.f / fmaxf(sqrtf(red[0]), EPSN);
  bf16_t* ob = out + (size_t)b * D * K;
  for (int i = tid; i < D * K; i += 256) ob[i] = (bf16_t)(vlad[i] * rn);
}

// --------------------------------------------------------------------------
// WMMA bf16 GEMM:  C(M,N) = A(M,K) * B(N,K)^T   (both K-contiguous rows)
// mode 0: C = acc + bias[n]      (optional f32 and/or bf16 output)
// mode 1: C = rowA2[m] + rowB2[n] - 2*acc   (pairwise sq-distance epilogue)
// Requires M%128==0, N%64==0, K%32==0.  256 threads = 8 waves (wave32).
// Double-buffered LDS; tile k+1 is staged (async global->LDS DMA when
// available) while tile k is consumed by WMMA. Fragments are 16-byte LDS
// chunks matching the CDNA5 16-bit A (K=(e&7)+8g+16*(e>>3)) and
// B (same K order as A) VGPR layouts.
// --------------------------------------------------------------------------
__global__ __launch_bounds__(256) void k_wmma_gemm(
    const bf16_t* __restrict__ A, const bf16_t* __restrict__ B,
    const float* __restrict__ bias, float* __restrict__ Cf,
    bf16_t* __restrict__ Cb, const float* __restrict__ rowA2,
    const float* __restrict__ rowB2, int M, int N, int K, int mode) {
  __shared__ bf16_t As[2][128][40];  // 80B row stride (16B-aligned chunks)
  __shared__ bf16_t Bs[2][64][40];
  const int tid = threadIdx.x;
  const int lane = tid & 31;
  const int wave = tid >> 5;  // 0..7
  const int wm = wave >> 1;   // 0..3  (M tile of wave)
  const int wn = wave & 1;    // 0..1  (N tile of wave)
  const int g = lane >> 4;    // lane group
  const int mr = lane & 15;
  const int m0 = blockIdx.y * 128;
  const int n0 = blockIdx.x * 64;

  v8f acc[2][2];
  for (int i = 0; i < 2; ++i)
    for (int j = 0; j < 2; ++j)
      for (int r = 0; r < 8; ++r) acc[i][j][r] = 0.f;

  const int arow = tid >> 1, ahalf = tid & 1;  // A: 128 rows x 2 halves
  const int brow = tid >> 2, bq = tid & 3;     // B: 64 rows x 4 chunks
  const int ksteps = K >> 5;

  // stage one 128x32 A tile + 64x32 B tile into LDS buffer `buf`
  auto stage_tile = [&](int kt, int buf) {
    const bf16_t* ag = A + (size_t)(m0 + arow) * K + (kt << 5) + (ahalf << 4);
    bf16_t* ad = &As[buf][arow][ahalf * 16];
    const bf16_t* bg = B + (size_t)(n0 + brow) * K + (kt << 5) + (bq << 3);
    bf16_t* bd = &Bs[buf][brow][bq * 8];
#if USE_ASYNC_LDS
    bf16_t* agm = const_cast<bf16_t*>(ag);
    bf16_t* bgm = const_cast<bf16_t*>(bg);
    __builtin_amdgcn_global_load_async_to_lds_b128((AS1 v4i*)agm,
                                                   (AS3 v4i*)ad, 0, 0);
    __builtin_amdgcn_global_load_async_to_lds_b128((AS1 v4i*)(agm + 8),
                                                   (AS3 v4i*)(ad + 8), 0, 0);
    __builtin_amdgcn_global_load_async_to_lds_b128((AS1 v4i*)bgm,
                                                   (AS3 v4i*)bd, 0, 0);
#else
    *(v8bf*)ad = *(const v8bf*)ag;
    *(v8bf*)(ad + 8) = *(const v8bf*)(ag + 8);
    *(v8bf*)bd = *(const v8bf*)bg;
    if (kt + 1 < ksteps)  // global_prefetch_b8 of the following A tile
      __builtin_prefetch(A + (size_t)(m0 + arow) * K + ((kt + 1) << 5), 0, 1);
#endif
  };

  stage_tile(0, 0);
#if USE_ASYNC_LDS
  __builtin_amdgcn_s_wait_asynccnt(0);
#endif
  __syncthreads();

  for (int kt = 0; kt < ksteps; ++kt) {
    const int cur = kt & 1;
    // Buffer cur^1 was last *read* in iteration kt-1; the barrier at the end
    // of that iteration makes it safe to overwrite now, overlapping with the
    // WMMA work on buffer cur below.
    if (kt + 1 < ksteps) stage_tile(kt + 1, cur ^ 1);

    union Frag { v16bf v; v8bf h[2]; } a[2], bfr[2];
    for (int i = 0; i < 2; ++i) {
      const bf16_t* p = &As[cur][wm * 32 + i * 16 + mr][0];
      a[i].h[0] = *(const v8bf*)(p + 8 * g);       // K = 8g..8g+7
      a[i].h[1] = *(const v8bf*)(p + 16 + 8 * g);  // K = 16+8g..23+8g
    }
    for (int j = 0; j < 2; ++j) {
      const bf16_t* p = &Bs[cur][wn * 32 + j * 16 + mr][0];
      bfr[j].h[0] = *(const v8bf*)(p + 8 * g);       // K = 8g..8g+7
      bfr[j].h[1] = *(const v8bf*)(p + 16 + 8 * g);  // K = 16+8g..23+8g
    }
    for (int i = 0; i < 2; ++i)
      for (int j = 0; j < 2; ++j)
        acc[i][j] = __builtin_amdgcn_wmma_f32_16x16x32_bf16(
            false, a[i].v, false, bfr[j].v, (short)0, acc[i][j], false, false);

#if USE_ASYNC_LDS
    __builtin_amdgcn_s_wait_asynccnt(0);  // own async writes to buf cur^1 done
#endif
    __syncthreads();  // publish LDS writes workgroup-wide
  }

  // C/D layout: lane<16 -> M=r, lane>=16 -> M=r+8 ; N = lane&15
  for (int i = 0; i < 2; ++i)
    for (int j = 0; j < 2; ++j) {
      int col = n0 + wn * 32 + j * 16 + mr;
      float badd = (mode == 0 && bias) ? bias[col] : 0.f;
      for (int r = 0; r < 8; ++r) {
        int row = m0 + wm * 32 + i * 16 + 8 * g + r;
        float v = acc[i][j][r];
        if (mode == 0)
          v += badd;
        else
          v = rowA2[row] + rowB2[col] - 2.0f * v;
        size_t oi = (size_t)row * N + col;
        if (Cf) Cf[oi] = v;
        if (Cb) Cb[oi] = (bf16_t)v;
      }
    }
}

// --------------------------------------------------------------------------
// Small fp32 GEMM (NN): C(M,N) = A(M,K) * B(K,N), bf16 output. Used once per
// call to fold fc2@fc1 -> W12. All dims multiples of 16 by construction.
// --------------------------------------------------------------------------
__global__ __launch_bounds__(256) void k_fold_gemm(
    const float* __restrict__ A, const float* __restrict__ B,
    bf16_t* __restrict__ C, int M, int N, int K) {
  __shared__ float As[16][17], Bs[16][17];
  int tx = threadIdx.x & 15, ty = threadIdx.x >> 4;
  int row = blockIdx.y * 16 + ty;
  int col = blockIdx.x * 16 + tx;
  float acc = 0.f;
  for (int k0 = 0; k0 < K; k0 += 16) {
    As[ty][tx] = A[(size_t)row * K + k0 + tx];
    Bs[ty][tx] = B[(size_t)(k0 + ty) * N + col];
    __syncthreads();
    for (int kk = 0; kk < 16; ++kk) acc += As[ty][kk] * Bs[kk][tx];
    __syncthreads();
  }
  C[(size_t)row * N + col] = (bf16_t)acc;
}

// bias fold: out[o] = b2[o] + sum_h w2[o,h]*b1[h]   (out dim = 256)
__global__ __launch_bounds__(256) void k_fold_bias(const float* __restrict__ w2,
                                                   const float* __restrict__ b1,
                                                   const float* __restrict__ b2,
                                                   float* __restrict__ out,
                                                   int hid) {
  int o = threadIdx.x;
  float s = b2[o];
  for (int h = 0; h < hid; ++h) s += w2[(size_t)o * hid + h] * b1[h];
  out[o] = s;
}

// --------------------------------------------------------------------------
// Context gating + L2 norm: e = l2norm(y2 * sigmoid(bn(gpre))) per row (256).
// Optionally emits bf16 copy and per-row sum-of-squares (for distance GEMM).
// --------------------------------------------------------------------------
__global__ __launch_bounds__(256) void k_gate(
    const float* __restrict__ y2, const float* __restrict__ gpre,
    const float* __restrict__ bnb, const float* __restrict__ bnm,
    const float* __restrict__ bns, const float* __restrict__ bnv,
    float* __restrict__ ef, bf16_t* __restrict__ eb,
    float* __restrict__ ss_out) {
  __shared__ float red[256];
  const int n = blockIdx.x, tid = threadIdx.x;
  size_t idx = (size_t)n * 256 + tid;
  float gv = gpre[idx];
  gv = (gv - bnm[tid]) * rsqrtf(bnv[tid] + 1e-5f) * bns[tid] + bnb[tid];
  float h = y2[idx] * (1.0f / (1.0f + __expf(-gv)));
  red[tid] = h * h;
  __syncthreads();
  for (int s = 128; s > 0; s >>= 1) {
    if (tid < s) red[tid] += red[tid + s];
    __syncthreads();
  }
  float rn = 1.f / fmaxf(sqrtf(red[0]), EPSN);
  float e = h * rn;
  ef[idx] = e;
  if (eb) eb[idx] = (bf16_t)e;
  if (ss_out && tid == 0) ss_out[n] = red[0] * rn * rn;
}

// --------------------------------------------------------------------------
// MoE gate: softmax(t @ moe_w^T + moe_b) * ind, renormalized.  t is bf16.
// --------------------------------------------------------------------------
__global__ __launch_bounds__(256) void k_moe(const bf16_t* __restrict__ t,
                                             const float* __restrict__ w,
                                             const float* __restrict__ bsc,
                                             const float* __restrict__ ind,
                                             float* __restrict__ moe) {
  __shared__ float red[256];
  const int n = blockIdx.x, tid = threadIdx.x;
  const int m = tid & 3, li = tid >> 2;  // 64 partials per expert
  const bf16_t* tb = t + (size_t)n * 9600;
  float s = 0.f;
  for (int i = li; i < 9600; i += 64) s += (float)tb[i] * w[m * 9600 + i];
  red[tid] = s;
  __syncthreads();
  for (int st = 128; st >= 4; st >>= 1) {  // stride stays multiple of 4
    if (tid < st) red[tid] += red[tid + st];
    __syncthreads();
  }
  if (tid == 0) {
    float l[4], mx = -3.0e38f;
    for (int j = 0; j < 4; ++j) {
      l[j] = red[j] + bsc[j];
      mx = fmaxf(mx, l[j]);
    }
    float sm = 0.f;
    for (int j = 0; j < 4; ++j) {
      l[j] = __expf(l[j] - mx);
      sm += l[j];
    }
    float inv = 1.f / sm, q = 0.f;
    for (int j = 0; j < 4; ++j) {
      l[j] = l[j] * inv * ind[(size_t)n * 4 + j];
      q += l[j];
    }
    float iq = 1.f / q;
    for (int j = 0; j < 4; ++j) moe[(size_t)n * 4 + j] = l[j] * iq;
  }
}

// --------------------------------------------------------------------------
// fuse[n,:] = sum_m moe[n,m] * emb_m[n,:]; emit f32 + bf16 + row sumsq.
// --------------------------------------------------------------------------
__global__ __launch_bounds__(256) void k_fuse(
    const float* __restrict__ moe, const float* __restrict__ ea,
    const float* __restrict__ efc, const float* __restrict__ em,
    const float* __restrict__ ev, float* __restrict__ ff,
    bf16_t* __restrict__ fb, float* __restrict__ f2) {
  __shared__ float red[256];
  const int n = blockIdx.x, tid = threadIdx.x;
  size_t idx = (size_t)n * 256 + tid;
  const float* mw = moe + (size_t)n * 4;
  float v = mw[0] * ea[idx] + mw[1] * efc[idx] + mw[2] * em[idx] +
            mw[3] * ev[idx];
  ff[idx] = v;
  fb[idx] = (bf16_t)v;
  red[tid] = v * v;
  __syncthreads();
  for (int s = 128; s > 0; s >>= 1) {
    if (tid < s) red[tid] += red[tid + s];
    __syncthreads();
  }
  if (tid == 0) f2[n] = red[0];
}

// ===========================================================================
// Host-side orchestration.
// Input order: text, audio, face, motion, visual, ind, then params flattened
// in JAX tree_leaves order (dict keys sorted recursively):
//   gu.{audio,face,motion,visual}.{b1,b2,bcg,bn.bias,bn.mean,bn.scale,bn.var,
//                                  w1,w2,wcg}            -> idx 6..45
//   moe_b=46, moe_w=47
//   text_gu.{same 10 leaves}                             -> idx 48..57
//   vlad_audio.{bn.bias,bn.mean,bn.scale,bn.var,clusters,clusters2} -> 58..63
//   vlad_text.{same}                                     -> idx 64..69
// ===========================================================================
extern "C" void kernel_launch(void* const* d_in, const int* in_sizes, int n_in,
                              void* d_out, int out_size, void* d_ws,
                              size_t ws_size, hipStream_t stream) {
  (void)in_sizes; (void)n_in; (void)out_size; (void)ws_size;
  const float* text = (const float*)d_in[0];
  const float* audio = (const float*)d_in[1];
  const float* face = (const float*)d_in[2];
  const float* motion = (const float*)d_in[3];
  const float* visual = (const float*)d_in[4];
  const float* ind = (const float*)d_in[5];
  auto P = [&](int i) { return (const float*)d_in[i]; };

  // GU order: audio, face, motion, visual, text
  const int GU_BASE[5] = {6, 16, 26, 36, 48};
  const int GU_IN[5] = {1024, 128, 2048, 2048, 9600};
  const int GU_HID[5] = {768, 512, 1024, 1024, 1024};
  // leaf offsets within a GU: b1=0,b2=1,bcg=2,bnb=3,bnm=4,bns=5,bnv=6,w1=7,
  //                           w2=8,wcg=9

  // -------- workspace carve --------
  char* base = (char*)d_ws;
  size_t off = 0;
  auto carve = [&](size_t bytes) -> char* {
    char* p = base + off;
    off += (bytes + 255) & ~(size_t)255;
    return p;
  };
  bf16_t* t_bf = (bf16_t*)carve((size_t)NB * 9600 * 2);
  bf16_t* av_bf = (bf16_t*)carve((size_t)NB * 1024 * 2);
  bf16_t* in_bf[5];
  in_bf[0] = av_bf;
  in_bf[1] = (bf16_t*)carve((size_t)NB * 128 * 2);
  in_bf[2] = (bf16_t*)carve((size_t)NB * 2048 * 2);
  in_bf[3] = (bf16_t*)carve((size_t)NB * 2048 * 2);
  in_bf[4] = t_bf;
  bf16_t* W12[5];
  bf16_t* wcgb[5];
  float* biasF[5];
  float* y2f[5];
  bf16_t* y2b[5];
  float* emb[5];
  for (int i = 0; i < 5; ++i) W12[i] = (bf16_t*)carve((size_t)256 * GU_IN[i] * 2);
  for (int i = 0; i < 5; ++i) wcgb[i] = (bf16_t*)carve((size_t)256 * 256 * 2);
  for (int i = 0; i < 5; ++i) biasF[i] = (float*)carve(256 * 4);
  for (int i = 0; i < 5; ++i) y2f[i] = (float*)carve((size_t)NB * 256 * 4);
  for (int i = 0; i < 5; ++i) y2b[i] = (bf16_t*)carve((size_t)NB * 256 * 2);
  float* gpre = (float*)carve((size_t)NB * 256 * 4);
  for (int i = 0; i < 5; ++i) emb[i] = (float*)carve((size_t)NB * 256 * 4);
  bf16_t* temb_bf = (bf16_t*)carve((size_t)NB * 256 * 2);
  float* t2 = (float*)carve((size_t)NB * 4);
  float* moe = (float*)carve((size_t)NB * 4 * 4);
  float* fusef = (float*)carve((size_t)NB * 256 * 4);
  bf16_t* fuseb = (bf16_t*)carve((size_t)NB * 256 * 2);
  float* f2 = (float*)carve((size_t)NB * 4);

  // -------- bf16 conversions of raw activations + gating weights --------
  k_cvt_bf16<<<512, 256, 0, stream>>>(face, in_bf[1], (long)NB * 128);
  k_cvt_bf16<<<2048, 256, 0, stream>>>(motion, in_bf[2], (long)NB * 2048);
  k_cvt_bf16<<<2048, 256, 0, stream>>>(visual, in_bf[3], (long)NB * 2048);
  for (int i = 0; i < 5; ++i)
    k_cvt_bf16<<<256, 256, 0, stream>>>(P(GU_BASE[i] + 9), wcgb[i],
                                        (long)256 * 256);

  // -------- NetVLAD (audio then text) --------
  k_netvlad<<<NB, 256, 0, stream>>>(audio, P(62), P(63), P(58), P(59), P(60),
                                    P(61), av_bf, 20, 128, 8);
  k_netvlad<<<NB, 256, 0, stream>>>(text, P(68), P(69), P(64), P(65), P(66),
                                    P(67), t_bf, 30, 300, 32);

  // -------- fold fc2@fc1 -> W12 (bf16) and fused bias --------
  for (int i = 0; i < 5; ++i) {
    int b = GU_BASE[i];
    dim3 gf(GU_IN[i] / 16, 256 / 16);
    k_fold_gemm<<<gf, 256, 0, stream>>>(P(b + 8), P(b + 7), W12[i], 256,
                                        GU_IN[i], GU_HID[i]);
    k_fold_bias<<<1, 256, 0, stream>>>(P(b + 8), P(b + 0), P(b + 1), biasF[i],
                                       GU_HID[i]);
  }

  // -------- per-GU: main WMMA GEMM, gating GEMM, gate epilogue --------
  dim3 g1(256 / 64, NB / 128);
  for (int i = 0; i < 5; ++i) {
    int b = GU_BASE[i];
    k_wmma_gemm<<<g1, 256, 0, stream>>>(in_bf[i], W12[i], biasF[i], y2f[i],
                                        y2b[i], nullptr, nullptr, NB, 256,
                                        GU_IN[i], 0);
    k_wmma_gemm<<<g1, 256, 0, stream>>>(y2b[i], wcgb[i], P(b + 2), gpre,
                                        nullptr, nullptr, nullptr, NB, 256,
                                        256, 0);
    bool isText = (i == 4);
    k_gate<<<NB, 256, 0, stream>>>(y2f[i], gpre, P(b + 3), P(b + 4), P(b + 5),
                                   P(b + 6), emb[i],
                                   isText ? temb_bf : nullptr,
                                   isText ? t2 : nullptr);
  }

  // -------- MoE gate + fusion --------
  k_moe<<<NB, 256, 0, stream>>>(t_bf, P(47), P(46), ind, moe);
  k_fuse<<<NB, 256, 0, stream>>>(moe, emb[0], emb[1], emb[2], emb[3], fusef,
                                 fuseb, f2);

  // -------- conf = ||t_i||^2 + ||f_j||^2 - 2 t_i . f_j  (WMMA, mode 1) ----
  dim3 gc(NB / 64, NB / 128);
  k_wmma_gemm<<<gc, 256, 0, stream>>>(temb_bf, fuseb, nullptr, (float*)d_out,
                                      nullptr, t2, f2, NB, NB, 256, 1);
}
